// VectorQuantizer_4432406249685
// MI455X (gfx1250) — matrix operand order repaired and checked by reading, not compiled, not run on hardware
//
#include <hip/hip_runtime.h>
#include <math.h>

typedef __attribute__((ext_vector_type(16))) __bf16 v16bf;
typedef __attribute__((ext_vector_type(8)))  float  v8f;

#define C_DIM   128
#define K_CODES 512
#define HW      1024
#define NUMEL   8388608   /* 64*128*32*32 */
#define NPOS    65536     /* 64*32*32 */

// ---------------------------------------------------------------------------
// Zero the workspace: 512 u32 histogram bins + 256 f32 per-workgroup SSE parts
// ---------------------------------------------------------------------------
__global__ void vq_init_kernel(unsigned* __restrict__ ws) {
  int t = blockIdx.x * blockDim.x + threadIdx.x;
  if (t < 768) ws[t] = 0u;
}

// ---------------------------------------------------------------------------
// Main fused kernel: bf16x3 WMMA GEMM (x · w^T) + argmin + gather + loss parts
// grid = 256 blocks of 256 threads (8 wave32). Block owns 256 positions of one
// image b = blockIdx.x/4. Codebook lives in LDS as bf16 hi/lo (256 KB).
// ---------------------------------------------------------------------------
__global__ __launch_bounds__(256) void vq_main_kernel(
    const float* __restrict__ inp, const float* __restrict__ w,
    float* __restrict__ out, unsigned* __restrict__ counts,
    float* __restrict__ sseParts)
{
  __shared__ __bf16 sWHi[K_CODES * C_DIM];   // 128 KB
  __shared__ __bf16 sWLo[K_CODES * C_DIM];   // 128 KB
  __shared__ float  sW2n[K_CODES];           // -0.5*||w_k||^2
  __shared__ int    sIdx[128];               // per-round winning code ids
  __shared__ float  sSse[8];                 // per-wave SSE partials

  const int tid  = threadIdx.x;
  const int lane = tid & 31;
  const int wv   = tid >> 5;
  const int b      = blockIdx.x >> 2;
  const int pstart = (blockIdx.x & 3) << 8;  // 256 positions per block

  if (tid < 8) sSse[tid] = 0.f;

  // ---- Phase 0: -0.5*||w_k||^2 in exact fp32 (2 codes / thread) ----
  for (int r = 0; r < 2; ++r) {
    int k = tid + 256 * r;
    const float4* wr = (const float4*)(w + k * C_DIM);
    float s = 0.f;
    #pragma unroll
    for (int i = 0; i < 32; ++i) {
      float4 f = wr[i];
      s += f.x * f.x + f.y * f.y + f.z * f.z + f.w * f.w;
    }
    sW2n[k] = -0.5f * s;
  }

  // ---- Phase 1: codebook fp32 -> bf16 hi/lo split into LDS ----
  for (int i = 0; i < 64; ++i) {
    int e4 = i * 256 + tid;                  // float4 index, coalesced
    float4 f = ((const float4*)w)[e4];
    int e = e4 * 4;
    float fv[4] = { f.x, f.y, f.z, f.w };
    #pragma unroll
    for (int j = 0; j < 4; ++j) {
      __bf16 h = (__bf16)fv[j];
      sWHi[e + j] = h;
      sWLo[e + j] = (__bf16)(fv[j] - (float)h);
    }
  }
  __syncthreads();

  const int half = lane >> 4;   // which K-half of the WMMA operand this lane holds
  const int col  = lane & 15;   // N (code) column / M (position) within tile

  for (int r = 0; r < 2; ++r) {
    const int pRound = pstart + r * 128;     // round base within image
    const int pbase  = pRound + wv * 16;     // this wave's 16 positions

    // ---- A fragments (16 pos x 128 ch), loaded once per round, split hi/lo ----
    v16bf aHi[4], aLo[4];
    {
      const float* xcol = inp + (size_t)b * (C_DIM * HW) + (pbase + col);
      #pragma unroll
      for (int q = 0; q < 4; ++q) {
        #pragma unroll
        for (int j = 0; j < 8; ++j) {
          // Contiguous K order chosen here (B matches): lanes<16 hold K = 32q+{0..15},
          // lanes>=16 hold K = 32q+{16..31}, packed 2 per VGPR.
          int c0 = 32 * q + 16 * half + (j << 1);
          float f0 = xcol[(size_t)c0 * HW];
          float f1 = xcol[(size_t)(c0 + 1) * HW];
          __bf16 h0 = (__bf16)f0, h1 = (__bf16)f1;
          aHi[q][2 * j + 0] = h0;
          aHi[q][2 * j + 1] = h1;
          aLo[q][2 * j + 0] = (__bf16)(f0 - (float)h0);
          aLo[q][2 * j + 1] = (__bf16)(f1 - (float)h1);
        }
      }
    }

    float bestVal[8];
    int   bestIdx[8];
    #pragma unroll
    for (int j = 0; j < 8; ++j) { bestVal[j] = -3.4e38f; bestIdx[j] = 0; }

    // ---- 32 code tiles x (4 K-chunks x 3 bf16 WMMAs) ----
    for (int t = 0; t < 32; ++t) {
      int k = t * 16 + col;
      float cinit = sW2n[k];                         // fold -0.5||w||^2 into C
      v8f acc  = { cinit, cinit, cinit, cinit, cinit, cinit, cinit, cinit };
      v8f acc2 = { 0.f, 0.f, 0.f, 0.f, 0.f, 0.f, 0.f, 0.f };
      const __bf16* rowH = sWHi + k * C_DIM + 16 * half;
      const __bf16* rowL = sWLo + k * C_DIM + 16 * half;
      #pragma unroll
      for (int q = 0; q < 4; ++q) {
        v16bf bH = *(const v16bf*)(rowH + 32 * q);   // 32B aligned LDS reads
        v16bf bL = *(const v16bf*)(rowL + 32 * q);
        acc  = __builtin_amdgcn_wmma_f32_16x16x32_bf16(false, aHi[q], false, bH,
                                                       (short)0, acc,  false, false);
        acc2 = __builtin_amdgcn_wmma_f32_16x16x32_bf16(false, aHi[q], false, bL,
                                                       (short)0, acc2, false, false);
        acc2 = __builtin_amdgcn_wmma_f32_16x16x32_bf16(false, aLo[q], false, bH,
                                                       (short)0, acc2, false, false);
      }
      #pragma unroll
      for (int j = 0; j < 8; ++j) {
        float s = acc[j] + acc2[j];                  // = x.w - 0.5||w||^2
        bool better = s > bestVal[j];                // argmax == argmin distance
        bestVal[j] = better ? s : bestVal[j];
        bestIdx[j] = better ? k : bestIdx[j];
      }
    }

    // ---- cross-lane argmax across the 16 code columns (wave32 butterfly) ----
    #pragma unroll
    for (int off = 1; off < 16; off <<= 1) {
      #pragma unroll
      for (int j = 0; j < 8; ++j) {
        float ov = __shfl_xor(bestVal[j], off, 32);
        int   oi = __shfl_xor(bestIdx[j], off, 32);
        if (ov > bestVal[j] || (ov == bestVal[j] && oi < bestIdx[j])) {
          bestVal[j] = ov; bestIdx[j] = oi;          // first-index tie-break
        }
      }
    }
    if (col == 0) {
      int basep = wv * 16 + 8 * half;                // D rows: lanes<16 -> M 0..7
      #pragma unroll
      for (int j = 0; j < 8; ++j) sIdx[basep + j] = bestIdx[j];
    }
    __syncthreads();

    // ---- gather codes (exact fp32), write output, SSE, histogram ----
    {
      int pl = tid & 127;
      int ch = tid >> 7;                             // channel half (0/1)
      int code = sIdx[pl];
      const float* wrow = w + code * C_DIM;
      size_t xoff = (size_t)b * (C_DIM * HW) + (size_t)(pRound + pl);
      float sse = 0.f;
      #pragma unroll
      for (int i = 0; i < 64; ++i) {
        int c = ch * 64 + i;
        float qv = wrow[c];
        size_t o = xoff + (size_t)c * HW;
        float xv = inp[o];                           // coalesced along p
        float dq = qv - xv;
        out[o] = xv + dq;                            // mirrors x + (q - x)
        sse += dq * dq;
      }
      if (ch == 0) atomicAdd(counts + code, 1u);     // integer: deterministic
      #pragma unroll
      for (int off = 16; off >= 1; off >>= 1) sse += __shfl_xor(sse, off, 32);
      if (lane == 0) sSse[wv] += sse;                // fixed order per wave
    }
    __syncthreads();
  }

  if (tid == 0) {
    float tot = 0.f;
    #pragma unroll
    for (int i = 0; i < 8; ++i) tot += sSse[i];      // fixed order
    sseParts[blockIdx.x] = tot;
  }
}

// ---------------------------------------------------------------------------
// Finalize: loss = 1.25 * SSE/NUMEL ; perplexity = exp(-sum p log(p+1e-10))
// ---------------------------------------------------------------------------
__global__ void vq_finalize_kernel(const unsigned* __restrict__ counts,
                                   const float* __restrict__ parts,
                                   float* __restrict__ tail)
{
  __shared__ float red[256];
  int t = threadIdx.x;
  red[t] = parts[t];
  __syncthreads();
  for (int off = 128; off > 0; off >>= 1) {
    if (t < off) red[t] += red[t + off];
    __syncthreads();
  }
  float sseTotal = red[0];
  __syncthreads();
  float e = 0.f;
  #pragma unroll
  for (int i = 0; i < 2; ++i) {
    float p = (float)counts[t + 256 * i] * (1.0f / (float)NPOS);
    e += p * logf(p + 1e-10f);
  }
  red[t] = e;
  __syncthreads();
  for (int off = 128; off > 0; off >>= 1) {
    if (t < off) red[t] += red[t + off];
    __syncthreads();
  }
  if (t == 0) {
    tail[0] = 1.25f * sseTotal * (1.0f / (float)NUMEL);
    tail[1] = expf(-red[0]);
  }
}

// ---------------------------------------------------------------------------
extern "C" void kernel_launch(void* const* d_in, const int* in_sizes, int n_in,
                              void* d_out, int out_size, void* d_ws, size_t ws_size,
                              hipStream_t stream) {
  const float* inp = (const float*)d_in[0];   // (64,128,32,32) fp32
  const float* w   = (const float*)d_in[1];   // (512,128) fp32
  float* out = (float*)d_out;                 // 8388608 + 2 floats

  unsigned* counts = (unsigned*)d_ws;         // [512] u32
  float* parts     = (float*)d_ws + 512;      // [256] f32

  vq_init_kernel<<<3, 256, 0, stream>>>((unsigned*)d_ws);
  vq_main_kernel<<<256, 256, 0, stream>>>(inp, w, out, counts, parts);
  vq_finalize_kernel<<<1, 256, 0, stream>>>(counts, parts, out + NUMEL);
}
